// IPAttnProcessor_60112362275307
// MI455X (gfx1250) — matrix operand order repaired and checked by reading, not compiled, not run on hardware
//
#include <hip/hip_runtime.h>
#include <hip/hip_bf16.h>
#include <math.h>

typedef __bf16 bf16_t;
typedef __attribute__((ext_vector_type(16))) __bf16 v16bf;
typedef __attribute__((ext_vector_type(8)))  __bf16 v8bf;
typedef __attribute__((ext_vector_type(4)))  __bf16 v4bf;
typedef __attribute__((ext_vector_type(8)))  float  v8f;
typedef __attribute__((ext_vector_type(4)))  float  v4f;
typedef __attribute__((ext_vector_type(4)))  int    v4i;

#define D_MODEL 2048
#define NHEADS  16
#define DH      128
#define TDIM    1280
#define BATCH   2
#define LQ      2048
#define LIP     512
#define MTOT    (LQ + LIP)            // 2560 keys total
#define QSCALE  0.08838834764831845f  // 1/sqrt(128)

// ---- CDNA5 async global->LDS path (guarded; falls back to load+ds_store) ----
#if __has_builtin(__builtin_amdgcn_global_load_async_to_lds_b128)
#define USE_ASYNC_LDS 1
typedef __attribute__((address_space(1))) v4i gv4i;   // global int4
typedef __attribute__((address_space(3))) v4i lv4i;   // LDS int4
static __device__ __forceinline__ gv4i* to_global(const void* p) {
    return (gv4i*)(unsigned long long)(size_t)p;
}
static __device__ __forceinline__ lv4i* to_local(const void* p) {
    // LDS aperture: low 32 bits of the generic address are the LDS offset
    return (lv4i*)(unsigned int)(size_t)p;
}
static __device__ __forceinline__ void wait_async0() {
#if __has_builtin(__builtin_amdgcn_s_wait_asynccnt)
    __builtin_amdgcn_s_wait_asynccnt(0);
#else
    asm volatile("s_wait_asynccnt 0" ::: "memory");
#endif
}
#endif

// ---------------------------------------------------------------------------
// 1) f32 -> bf16 weight conversion (W_k_ip, W_v_ip)
// ---------------------------------------------------------------------------
__global__ void cvt_w_kernel(const float* __restrict__ wk, const float* __restrict__ wv,
                             bf16_t* __restrict__ wkb, bf16_t* __restrict__ wvb) {
    int i = blockIdx.x * blockDim.x + threadIdx.x;           // one thread = 4 elems
    const float* src = blockIdx.y ? wv : wk;
    bf16_t*      dst = blockIdx.y ? wvb : wkb;
    v4f x = *(const v4f*)(src + (size_t)i * 4);
    v4bf o;
#pragma unroll
    for (int j = 0; j < 4; ++j) o[j] = (bf16_t)x[j];
    *(v4bf*)(dst + (size_t)i * 4) = o;
}

// ---------------------------------------------------------------------------
// 2) adaLN: emb[b][j] = sum_t silu(t_emb[b][t]) * W_ada[j][t] + b_ada[j]
// ---------------------------------------------------------------------------
__global__ void ada_emb_kernel(const float* __restrict__ t_emb, const float* __restrict__ W_ada,
                               const float* __restrict__ b_ada, float* __restrict__ emb) {
    __shared__ float st[TDIM];
    int b = blockIdx.y;
    for (int t = threadIdx.x; t < TDIM; t += blockDim.x) {
        float x = t_emb[b * TDIM + t];
        st[t] = x / (1.0f + __expf(-x));   // silu
    }
    __syncthreads();
    int j = blockIdx.x * blockDim.x + threadIdx.x;           // 0..4095
    const float* wrow = W_ada + (size_t)j * TDIM;
    float acc = 0.0f;
    for (int t = 0; t < TDIM; ++t) acc = fmaf(st[t], wrow[t], acc);
    emb[(size_t)b * 2 * D_MODEL + j] = acc + b_ada[j];
}

// ---------------------------------------------------------------------------
// 3) LayerNorm + (1+scale)/shift of ip_hidden_states -> bf16 ip_n
// ---------------------------------------------------------------------------
__global__ void ln_kernel(const float* __restrict__ xin, const float* __restrict__ emb,
                          bf16_t* __restrict__ ipn) {
    int row = blockIdx.x;                  // 0 .. B*LIP-1
    int b   = row / LIP;
    const float* x = xin + (size_t)row * D_MODEL;
    float xv[8], s = 0.f, s2 = 0.f;
#pragma unroll
    for (int k = 0; k < 8; ++k) {
        xv[k] = x[threadIdx.x + 256 * k];
        s  += xv[k];
        s2 += xv[k] * xv[k];
    }
#pragma unroll
    for (int off = 16; off; off >>= 1) {
        s  += __shfl_xor(s,  off, 32);
        s2 += __shfl_xor(s2, off, 32);
    }
    __shared__ float rs[8], rs2[8];
    __shared__ float mu_s, rv_s;
    int w = threadIdx.x >> 5;
    if ((threadIdx.x & 31) == 0) { rs[w] = s; rs2[w] = s2; }
    __syncthreads();
    if (threadIdx.x == 0) {
        float S = 0.f, S2 = 0.f;
#pragma unroll
        for (int i = 0; i < 8; ++i) { S += rs[i]; S2 += rs2[i]; }
        float mu  = S * (1.0f / D_MODEL);
        float var = S2 * (1.0f / D_MODEL) - mu * mu;
        mu_s = mu;
        rv_s = rsqrtf(var + 1e-6f);
    }
    __syncthreads();
    float mu = mu_s, rv = rv_s;
    bf16_t* dst = ipn + (size_t)row * D_MODEL;
    const float* sh = emb + (size_t)b * 2 * D_MODEL;
    const float* sc = sh + D_MODEL;
#pragma unroll
    for (int k = 0; k < 8; ++k) {
        int c = threadIdx.x + 256 * k;
        float y = (xv[k] - mu) * rv * (1.0f + sc[c]) + sh[c];
        dst[c] = (bf16_t)y;
    }
}

// ---------------------------------------------------------------------------
// 4) WMMA GEMM: C[m][n] = sum_k ipn[m][k] * W[n][k]    (M=1024, N=2048, K=2048)
//    one wave computes a 16x64 tile; A/B fragments both use contiguous k = 16*h + s order
// ---------------------------------------------------------------------------
__global__ void __launch_bounds__(256)
ipkv_gemm_kernel(const bf16_t* __restrict__ ipn, const bf16_t* __restrict__ wkb,
                 const bf16_t* __restrict__ wvb, float* __restrict__ ipk,
                 float* __restrict__ ipv) {
    const bf16_t* W = blockIdx.z ? wvb : wkb;
    float*        C = blockIdx.z ? ipv : ipk;
    int wave = threadIdx.x >> 5, lane = threadIdx.x & 31;
    int la = lane & 15, lh = lane >> 4;
    int m0 = (blockIdx.y * 8 + wave) * 16;
    int n0 = blockIdx.x * 64;
    int kha = lh * 16;   // A-frag K base within 32-chunk
    int khb = lh * 16;   // B-frag K base within 32-chunk

    v8f zero8 = {};
    v8f acc[4];
#pragma unroll
    for (int nt = 0; nt < 4; ++nt) acc[nt] = zero8;

    const bf16_t* arow = ipn + (size_t)(m0 + la) * D_MODEL;
    for (int k0 = 0; k0 < D_MODEL; k0 += 32) {
        v16bf a;
        v8bf alo = *(const v8bf*)(arow + k0 + kha);
        v8bf ahi = *(const v8bf*)(arow + k0 + kha + 8);
#pragma unroll
        for (int i = 0; i < 8; ++i) { a[i] = alo[i]; a[i + 8] = ahi[i]; }
#pragma unroll
        for (int nt = 0; nt < 4; ++nt) {
            const bf16_t* brow = W + (size_t)(n0 + nt * 16 + la) * D_MODEL + k0 + khb;
            v16bf bb = *(const v16bf*)brow;
            acc[nt] = __builtin_amdgcn_wmma_f32_16x16x32_bf16(
                false, a, false, bb, (short)0, acc[nt], false, false);
        }
    }
#pragma unroll
    for (int nt = 0; nt < 4; ++nt)
#pragma unroll
        for (int r = 0; r < 8; ++r) {
            int m = m0 + r + 8 * lh;
            int n = n0 + nt * 16 + la;
            C[(size_t)m * D_MODEL + n] = acc[nt][r];
        }
}

// ---------------------------------------------------------------------------
// 5a) RMS-norm pack (q / k / k_ip) -> bf16, head-major row-major (rows x 128)
// ---------------------------------------------------------------------------
__global__ void pack_rms_kernel(const float* __restrict__ src, const float* __restrict__ w,
                                bf16_t* __restrict__ dst, int Lsrc, int dstRows, int rowOff,
                                float outScale) {
    int wave = threadIdx.x >> 5, lane = threadIdx.x & 31;
    int rid = blockIdx.x * 8 + wave;                 // (b*Lsrc + l)*H + h
    int h = rid % NHEADS;
    int l = (rid / NHEADS) % Lsrc;
    int b = rid / (NHEADS * Lsrc);
    const float* x = src + ((size_t)(b * Lsrc + l) * D_MODEL) + h * DH;
    v4f xv = *(const v4f*)(x + lane * 4);
    float ss = xv[0] * xv[0] + xv[1] * xv[1] + xv[2] * xv[2] + xv[3] * xv[3];
#pragma unroll
    for (int off = 16; off; off >>= 1) ss += __shfl_xor(ss, off, 32);
    float rsig = rsqrtf(ss * (1.0f / DH) + 1e-6f) * outScale;
    v4f wv = *(const v4f*)(w + lane * 4);
    v4bf o;
#pragma unroll
    for (int j = 0; j < 4; ++j) o[j] = (bf16_t)(xv[j] * rsig * wv[j]);
    bf16_t* drow = dst + ((size_t)((b * NHEADS + h) * dstRows + rowOff + l)) * DH + lane * 4;
    *(v4bf*)drow = o;
}

// ---------------------------------------------------------------------------
// 5b) pack V transposed: vT[(b,h)][d][m]  (Dh x MTOT), no norm
// ---------------------------------------------------------------------------
__global__ void pack_vT_kernel(const float* __restrict__ src, bf16_t* __restrict__ dstT,
                               int Lsrc, int colOff) {
    int wave = threadIdx.x >> 5, lane = threadIdx.x & 31;
    int rid = blockIdx.x * 8 + wave;
    int h = rid % NHEADS;
    int l = (rid / NHEADS) % Lsrc;
    int b = rid / (NHEADS * Lsrc);
    const float* x = src + ((size_t)(b * Lsrc + l) * D_MODEL) + h * DH;
    v4f xv = *(const v4f*)(x + lane * 4);
    bf16_t* base = dstT + (size_t)(b * NHEADS + h) * DH * MTOT;
    int col = colOff + l;
#pragma unroll
    for (int j = 0; j < 4; ++j)
        base[(size_t)(lane * 4 + j) * MTOT + col] = (bf16_t)xv[j];
}

// ---------------------------------------------------------------------------
// 6) Flash attention with bf16 WMMA.
//    Per block: 8 waves x 16 query rows share one (b,h). K/V tiles (32 keys)
//    are staged once per block into LDS (async global->LDS when available),
//    eliminating the 8x redundant VMEM fetch across waves.
// ---------------------------------------------------------------------------
__global__ void __launch_bounds__(256)
attn_kernel(const bf16_t* __restrict__ qn, const bf16_t* __restrict__ kcat,
            const bf16_t* __restrict__ vT, float* __restrict__ out) {
    __shared__ __align__(16) bf16_t ktile[32][DH];     // 8 KB: keys x Dh
    __shared__ __align__(16) bf16_t vtile[DH][32];     // 8 KB: Dh x keys
    __shared__ __align__(16) bf16_t pbuf[8][16][32];   // 8 KB: wave-private P

    int tid  = threadIdx.x;
    int wave = tid >> 5, lane = tid & 31;
    int la = lane & 15, lh = lane >> 4;
    int b = blockIdx.z, h = blockIdx.y;
    int l0 = blockIdx.x * 128 + wave * 16;
    size_t bh = (size_t)(b * NHEADS + h);
    int kha = lh * 16, khb = lh * 16;

    // Q fragments for all of K-dim (4 chunks of 32), loaded once
    const bf16_t* qrow = qn + (bh * LQ + (size_t)(l0 + la)) * DH;
    v16bf qf[4];
#pragma unroll
    for (int kc = 0; kc < 4; ++kc) {
        v8bf lo = *(const v8bf*)(qrow + kc * 32 + kha);
        v8bf hi = *(const v8bf*)(qrow + kc * 32 + kha + 8);
#pragma unroll
        for (int i = 0; i < 8; ++i) { qf[kc][i] = lo[i]; qf[kc][i + 8] = hi[i]; }
    }

    v8f zero8 = {};
    v8f acc[8];                                    // 16 x 128 f32 output tile
#pragma unroll
    for (int d = 0; d < 8; ++d) acc[d] = zero8;
    float mi[8], li[8];
#pragma unroll
    for (int r = 0; r < 8; ++r) { mi[r] = -1e30f; li[r] = 0.0f; }

    const bf16_t* kgbase = kcat + bh * (size_t)MTOT * DH;
    const bf16_t* vgbase = vT + bh * (size_t)DH * MTOT;

    for (int kt = 0; kt < MTOT; kt += 32) {
        // ---- stage K (contiguous 8KB) and V (128 rows x 64B) into LDS ----
        {
            const char* kg = (const char*)(kgbase + (size_t)kt * DH);
            char*       ks = (char*)&ktile[0][0];
#ifdef USE_ASYNC_LDS
#pragma unroll
            for (int c = tid; c < 512; c += 256)
                __builtin_amdgcn_global_load_async_to_lds_b128(
                    to_global(kg + c * 16), to_local(ks + c * 16), 0, 0);
#pragma unroll
            for (int c = tid; c < 512; c += 256) {
                int row = c >> 2, sub = c & 3;
                __builtin_amdgcn_global_load_async_to_lds_b128(
                    to_global((const char*)(vgbase + (size_t)row * MTOT + kt) + sub * 16),
                    to_local((char*)&vtile[row][0] + sub * 16), 0, 0);
            }
            wait_async0();
#else
#pragma unroll
            for (int c = tid; c < 512; c += 256) {
                v8bf t = *(const v8bf*)(kg + c * 16);
                *(v8bf*)(ks + c * 16) = t;
            }
#pragma unroll
            for (int c = tid; c < 512; c += 256) {
                int row = c >> 2, sub = c & 3;
                v8bf t = *(const v8bf*)((const char*)(vgbase + (size_t)row * MTOT + kt) + sub * 16);
                *(v8bf*)((char*)&vtile[row][0] + sub * 16) = t;
            }
#endif
            // prefetch next tile's K rows while this tile computes
            if (kt + 32 < MTOT)
                __builtin_prefetch((const char*)(kgbase + (size_t)(kt + 32) * DH) + tid * 32, 0, 1);
        }
        __syncthreads();

        // ---- scores: two 16x16 tiles over K=128 (4 WMMAs each), from LDS ----
        v8f s0 = zero8, s1 = zero8;
#pragma unroll
        for (int kc = 0; kc < 4; ++kc) {
            v16bf bb0 = *(const v16bf*)&ktile[la][kc * 32 + khb];
            v16bf bb1 = *(const v16bf*)&ktile[16 + la][kc * 32 + khb];
            s0 = __builtin_amdgcn_wmma_f32_16x16x32_bf16(false, qf[kc], false, bb0,
                                                         (short)0, s0, false, false);
            s1 = __builtin_amdgcn_wmma_f32_16x16x32_bf16(false, qf[kc], false, bb1,
                                                         (short)0, s1, false, false);
        }
        // ---- online softmax over the 32 new keys ----
#pragma unroll
        for (int r = 0; r < 8; ++r) {
            float t = fmaxf(s0[r], s1[r]);
            t = fmaxf(t, __shfl_xor(t, 1, 32));
            t = fmaxf(t, __shfl_xor(t, 2, 32));
            t = fmaxf(t, __shfl_xor(t, 4, 32));
            t = fmaxf(t, __shfl_xor(t, 8, 32));
            float mnew  = fmaxf(mi[r], t);
            float alpha = __expf(mi[r] - mnew);
            float p0 = __expf(s0[r] - mnew);
            float p1 = __expf(s1[r] - mnew);
            float rs = p0 + p1;
            rs += __shfl_xor(rs, 1, 32);
            rs += __shfl_xor(rs, 2, 32);
            rs += __shfl_xor(rs, 4, 32);
            rs += __shfl_xor(rs, 8, 32);
            li[r] = li[r] * alpha + rs;
            mi[r] = mnew;
#pragma unroll
            for (int d = 0; d < 8; ++d) acc[d][r] *= alpha;
            int rowp = r + 8 * lh;                 // C-layout row of this register
            pbuf[wave][rowp][la]      = (bf16_t)p0;
            pbuf[wave][rowp][la + 16] = (bf16_t)p1;
        }
        __syncthreads();
        // ---- reshape P (C-layout -> A-frag) via LDS ----
        v16bf pf;
        {
            v8bf lo = *(const v8bf*)&pbuf[wave][la][kha];
            v8bf hi = *(const v8bf*)&pbuf[wave][la][kha + 8];
#pragma unroll
            for (int i = 0; i < 8; ++i) { pf[i] = lo[i]; pf[i + 8] = hi[i]; }
        }
        // ---- P x V: 8 WMMAs (one per 16-wide d chunk), V from LDS ----
#pragma unroll
        for (int d = 0; d < 8; ++d) {
            v16bf vb = *(const v16bf*)&vtile[d * 16 + la][khb];
            acc[d] = __builtin_amdgcn_wmma_f32_16x16x32_bf16(false, pf, false, vb,
                                                             (short)0, acc[d], false, false);
        }
        __syncthreads();   // protect ktile/vtile/pbuf before next iteration
    }

    // ---- epilogue: normalize by li and store (B, L, H*Dh) ----
#pragma unroll
    for (int r = 0; r < 8; ++r) {
        float inv = 1.0f / li[r];
        int row = l0 + r + 8 * lh;
        float* orow = out + ((size_t)(b * LQ + row)) * D_MODEL + h * DH;
#pragma unroll
        for (int d = 0; d < 8; ++d) orow[d * 16 + la] = acc[d][r] * inv;
    }
}

// ---------------------------------------------------------------------------
// launch
// ---------------------------------------------------------------------------
extern "C" void kernel_launch(void* const* d_in, const int* in_sizes, int n_in,
                              void* d_out, int out_size, void* d_ws, size_t ws_size,
                              hipStream_t stream) {
    (void)in_sizes; (void)n_in; (void)out_size; (void)ws_size;
    const float* ip_hidden = (const float*)d_in[0];
    const float* img_q     = (const float*)d_in[1];
    const float* img_k     = (const float*)d_in[2];
    const float* img_v     = (const float*)d_in[3];
    const float* t_emb     = (const float*)d_in[4];
    const float* W_ada     = (const float*)d_in[5];
    const float* b_ada     = (const float*)d_in[6];
    const float* W_k_ip    = (const float*)d_in[7];
    const float* W_v_ip    = (const float*)d_in[8];
    const float* w_q       = (const float*)d_in[9];
    const float* w_k       = (const float*)d_in[10];
    const float* w_ipk     = (const float*)d_in[11];
    float* out = (float*)d_out;

    char* ws = (char*)d_ws;
    size_t off = 0;
    auto carve = [&](size_t bytes) {
        void* p = ws + off;
        off += (bytes + 255) & ~(size_t)255;
        return p;
    };
    float*  emb  = (float*)carve((size_t)BATCH * 2 * D_MODEL * sizeof(float));
    bf16_t* wkb  = (bf16_t*)carve((size_t)D_MODEL * D_MODEL * 2);
    bf16_t* wvb  = (bf16_t*)carve((size_t)D_MODEL * D_MODEL * 2);
    bf16_t* ipn  = (bf16_t*)carve((size_t)BATCH * LIP * D_MODEL * 2);
    float*  ipk  = (float*)carve((size_t)BATCH * LIP * D_MODEL * sizeof(float));
    float*  ipv  = (float*)carve((size_t)BATCH * LIP * D_MODEL * sizeof(float));
    bf16_t* qnb  = (bf16_t*)carve((size_t)BATCH * NHEADS * LQ * DH * 2);
    bf16_t* kcat = (bf16_t*)carve((size_t)BATCH * NHEADS * MTOT * DH * 2);
    bf16_t* vT   = (bf16_t*)carve((size_t)BATCH * NHEADS * DH * MTOT * 2);

    // 1) weights -> bf16
    cvt_w_kernel<<<dim3(4096, 2), 256, 0, stream>>>(W_k_ip, W_v_ip, wkb, wvb);
    // 2) adaLN embedding
    ada_emb_kernel<<<dim3(16, BATCH), 256, 0, stream>>>(t_emb, W_ada, b_ada, emb);
    // 3) layernorm + scale/shift -> bf16 ip_n
    ln_kernel<<<BATCH * LIP, 256, 0, stream>>>(ip_hidden, emb, ipn);
    // 4) ip_key / ip_value GEMMs (WMMA)
    ipkv_gemm_kernel<<<dim3(32, 8, 2), 256, 0, stream>>>(ipn, wkb, wvb, ipk, ipv);
    // 5) pack q/k/v (+ ip parts) into WMMA-friendly bf16 layouts
    pack_rms_kernel<<<(BATCH * LQ * NHEADS) / 8, 256, 0, stream>>>(img_q, w_q, qnb, LQ, LQ, 0, QSCALE);
    pack_rms_kernel<<<(BATCH * LQ * NHEADS) / 8, 256, 0, stream>>>(img_k, w_k, kcat, LQ, MTOT, 0, 1.0f);
    pack_rms_kernel<<<(BATCH * LIP * NHEADS) / 8, 256, 0, stream>>>(ipk, w_ipk, kcat, LIP, MTOT, LQ, 1.0f);
    pack_vT_kernel<<<(BATCH * LQ * NHEADS) / 8, 256, 0, stream>>>(img_v, vT, LQ, 0);
    pack_vT_kernel<<<(BATCH * LIP * NHEADS) / 8, 256, 0, stream>>>(ipv, vT, LIP, LQ);
    // 6) flash attention (WMMA + async LDS staging)
    attn_kernel<<<dim3(LQ / 128, NHEADS, BATCH), 256, 0, stream>>>(qnb, kcat, vT, out);
}
